// GraphConnectionBlock_1434519077336
// MI455X (gfx1250) — compile-verified, matrix operand order checked by reading, not run on hardware
//
#include <hip/hip_runtime.h>
#include <hip/hip_fp16.h>
#include <stdint.h>

typedef __attribute__((ext_vector_type(16))) _Float16 v16h;
typedef __attribute__((ext_vector_type(8)))  _Float16 v8h;
typedef __attribute__((ext_vector_type(8)))  float    v8f;

// ---------------------------------------------------------------------------
// Exact GELU (approximate=False): 0.5*x*(1+erf(x/sqrt(2)))
__device__ __forceinline__ float gelu_exact(float x) {
    return 0.5f * x * (1.0f + erff(x * 0.70710678118654752440f));
}

// A-fragment loader for V_WMMA_F32_16X16X32_F16 from an LDS tile (row-major,
// `ld` halves between rows).  Layout per ISA 7.12.2 (16-bit A 16x32):
//   lanes 0-15 : row M = lane,   halves 0-7 = K0..K0+7, halves 8-15 = K0+16..23
//   lanes 16-31: row M = lane-16, same but K offset +8
__device__ __forceinline__ v16h load_a_frag(const _Float16* X, int ld, int k0, int lane) {
    int row = lane & 15;
    int kb  = k0 + ((lane & 16) ? 8 : 0);
    const _Float16* p = X + row * ld + kb;
    v8h lo = *(const v8h*)(p);
    v8h hi = *(const v8h*)(p + 16);
    v16h r;
#pragma unroll
    for (int i = 0; i < 8; ++i) { r[i] = lo[i]; r[i + 8] = hi[i]; }
    return r;
}

// ---------------------------------------------------------------------------
// Pack f32 weight matrix W[K][N] (row-major, [in,out]) into per-lane f16
// B-fragments: out[((nt*kTiles+kt)*32 + lane)*16 + h]
__global__ void pack_weights_kernel(const float* __restrict__ w,
                                    _Float16* __restrict__ out, int K, int N) {
    int kTiles = K >> 5;
    int total  = (N >> 4) * kTiles * 32 * 16;
    for (int idx = blockIdx.x * blockDim.x + threadIdx.x; idx < total;
         idx += gridDim.x * blockDim.x) {
        int h    = idx & 15;
        int lane = (idx >> 4) & 31;
        int t    = idx >> 9;
        int kt   = t % kTiles;
        int nt   = t / kTiles;
        int n    = nt * 16 + (lane & 15);
        int k    = kt * 32 + ((lane >> 4) & 1) * 8 + (h >> 3) * 16 + (h & 7);
        out[idx] = (_Float16)w[(size_t)k * N + n];
    }
}

__global__ void zero_u32_kernel(uint32_t* __restrict__ p, long n) {
    long i = (long)blockIdx.x * blockDim.x + threadIdx.x;
    long stride = (long)gridDim.x * blockDim.x;
    for (; i < n; i += stride) p[i] = 0u;
}

// ---------------------------------------------------------------------------
// Edge MLP: workgroup (4 waves) owns a 64-edge M-tile shared in LDS.
// Each wave covers a disjoint nt range; every B fragment loaded from L2 is
// reused by 4 WMMAs (one per 16-row M-subtile) -> 4x less L2 weight traffic.
#define ELD 392  // 384 + 8 pad (bank spread)
__global__ __launch_bounds__(128) void edge_mlp_kernel(
    const float* __restrict__ node, const float* __restrict__ edge,
    const int* __restrict__ edgeIdx,
    const float* __restrict__ g1, const float* __restrict__ b1,
    const _Float16* __restrict__ w1p, const float* __restrict__ bias1,
    const _Float16* __restrict__ w2p, const float* __restrict__ bias2,
    _Float16* __restrict__ Eh, float* __restrict__ edge_out, int nEdges) {
    __shared__ __align__(32) _Float16 sX[64 * ELD];   // ~49 KB
    __shared__ __align__(32) _Float16 sH[64 * ELD];   // ~49 KB
    const int lane = threadIdx.x & 31;
    const int wave = threadIdx.x >> 5;
    const long e0  = (long)blockIdx.x * 64;

    // ---- build + LayerNorm: wave builds rows [wave*16, wave*16+16) ----
    for (int r = wave * 16; r < wave * 16 + 16; ++r) {
        long e = e0 + r; if (e >= nEdges) e = nEdges - 1;
        int i0 = edgeIdx[2 * e];
        int i1 = edgeIdx[2 * e + 1];
        float x[12], s = 0.f, ss = 0.f;
#pragma unroll
        for (int j = 0; j < 12; ++j) {
            int col = lane + 32 * j;
            float v;
            if (col < 128)      v = edge[e * 128 + col];
            else if (col < 256) v = node[(long)i0 * 128 + (col - 128)];
            else                v = node[(long)i1 * 128 + (col - 256)];
            x[j] = v; s += v; ss += v * v;
        }
#pragma unroll
        for (int m = 16; m; m >>= 1) { s += __shfl_xor(s, m, 32); ss += __shfl_xor(ss, m, 32); }
        float mu  = s * (1.f / 384.f);
        float var = ss * (1.f / 384.f) - mu * mu;
        float rs  = rsqrtf(var + 1e-5f);
#pragma unroll
        for (int j = 0; j < 12; ++j) {
            int col = lane + 32 * j;
            sX[r * ELD + col] = (_Float16)((x[j] - mu) * rs * g1[col] + b1[col]);
        }
    }
    __syncthreads();

    // ---- GEMM1: [64x384] @ [384x384] + bias, GELU -> sH ----
    // wave handles nt in [wave*6, wave*6+6); 4-way M reuse per B fragment.
    const v16h* w1f = (const v16h*)w1p;
    for (int nti = 0; nti < 6; ++nti) {
        int nt = wave * 6 + nti;
        if (nti + 1 < 6) __builtin_prefetch((const void*)&w1f[((nt + 1) * 12) * 32 + lane], 0, 0);
        v8f acc[4] = {{}, {}, {}, {}};
#pragma unroll
        for (int kt = 0; kt < 12; ++kt) {
            v16h b = w1f[(nt * 12 + kt) * 32 + lane];
#pragma unroll
            for (int m = 0; m < 4; ++m) {
                v16h a = load_a_frag(sX + m * 16 * ELD, ELD, kt * 32, lane);
                acc[m] = __builtin_amdgcn_wmma_f32_16x16x32_f16(false, a, false, b, (short)0, acc[m], false, false);
            }
        }
        int col   = nt * 16 + (lane & 15);
        int rbase = (lane & 16) ? 8 : 0;
        float bv  = bias1[col];
#pragma unroll
        for (int m = 0; m < 4; ++m)
#pragma unroll
            for (int g = 0; g < 8; ++g)
                sH[(m * 16 + rbase + g) * ELD + col] = (_Float16)gelu_exact(acc[m][g] + bv);
    }
    __syncthreads();

    // ---- GEMM2: [64x384] @ [384x128] + bias -> Eh, edge_out ----
    // wave handles nt in [wave*2, wave*2+2)
    const v16h* w2f = (const v16h*)w2p;
    for (int nti = 0; nti < 2; ++nti) {
        int nt = wave * 2 + nti;
        v8f acc[4] = {{}, {}, {}, {}};
#pragma unroll
        for (int kt = 0; kt < 12; ++kt) {
            v16h b = w2f[(nt * 12 + kt) * 32 + lane];
#pragma unroll
            for (int m = 0; m < 4; ++m) {
                v16h a = load_a_frag(sH + m * 16 * ELD, ELD, kt * 32, lane);
                acc[m] = __builtin_amdgcn_wmma_f32_16x16x32_f16(false, a, false, b, (short)0, acc[m], false, false);
            }
        }
        int col   = nt * 16 + (lane & 15);
        int rbase = (lane & 16) ? 8 : 0;
        float bv  = bias2[col];
#pragma unroll
        for (int m = 0; m < 4; ++m)
#pragma unroll
            for (int g = 0; g < 8; ++g) {
                long e = e0 + m * 16 + rbase + g;
                if (e < nEdges) {
                    _Float16 hv = (_Float16)(acc[m][g] + bv);
                    Eh[e * 128 + col]       = hv;
                    edge_out[e * 128 + col] = edge[e * 128 + col] + (float)hv;
                }
            }
    }
}

// ---------------------------------------------------------------------------
// COO scatter-add in fp16: agg[row] += Eh[col] * val   (packed f16 atomics)
__global__ void scatter_kernel(const _Float16* __restrict__ Eh,
                               const int* __restrict__ rows,
                               const int* __restrict__ cols,
                               const float* __restrict__ vals,
                               _Float16* __restrict__ agg, int nnz) {
    long gid = (long)blockIdx.x * blockDim.x + threadIdx.x;
    int  i   = (int)(gid >> 6);   // 64 half2 pairs per nnz entry
    int  c2  = (int)(gid & 63);
    if (i >= nnz) return;
    int   col = cols[i], row = rows[i];
    float v   = vals[i];
    const _Float16* src = Eh + (size_t)col * 128 + c2 * 2;
    __half2 h2 = __floats2half2_rn((float)src[0] * v, (float)src[1] * v);
    atomicAdd((__half2*)(agg + (size_t)row * 128 + c2 * 2), h2);
}

// ---------------------------------------------------------------------------
// Node MLP: workgroup (4 waves) owns a 64-node M-tile; same 4x B reuse.
#define NLD 264  // 256 + 8 pad
__global__ __launch_bounds__(128) void node_mlp_kernel(
    const float* __restrict__ node, const _Float16* __restrict__ agg,
    const float* __restrict__ g2, const float* __restrict__ b2,
    const _Float16* __restrict__ w1p, const float* __restrict__ bias1,
    const _Float16* __restrict__ w2p, const float* __restrict__ bias2,
    float* __restrict__ node_out, int nNodes) {
    __shared__ __align__(32) _Float16 sX[64 * NLD];   // ~33 KB
    __shared__ __align__(32) _Float16 sH[64 * NLD];   // ~33 KB
    const int lane = threadIdx.x & 31;
    const int wave = threadIdx.x >> 5;
    const long n0  = (long)blockIdx.x * 64;

    for (int r = wave * 16; r < wave * 16 + 16; ++r) {
        long n = n0 + r; if (n >= nNodes) n = nNodes - 1;
        float x[8], s = 0.f, ss = 0.f;
#pragma unroll
        for (int j = 0; j < 8; ++j) {
            int col = lane + 32 * j;
            float v = (col < 128) ? node[n * 128 + col]
                                  : (float)agg[n * 128 + (col - 128)];
            x[j] = v; s += v; ss += v * v;
        }
#pragma unroll
        for (int m = 16; m; m >>= 1) { s += __shfl_xor(s, m, 32); ss += __shfl_xor(ss, m, 32); }
        float mu  = s * (1.f / 256.f);
        float var = ss * (1.f / 256.f) - mu * mu;
        float rs  = rsqrtf(var + 1e-5f);
#pragma unroll
        for (int j = 0; j < 8; ++j) {
            int col = lane + 32 * j;
            sX[r * NLD + col] = (_Float16)((x[j] - mu) * rs * g2[col] + b2[col]);
        }
    }
    __syncthreads();

    // GEMM1: 16 nt over 4 waves -> 4 each
    const v16h* w1f = (const v16h*)w1p;
    for (int nti = 0; nti < 4; ++nti) {
        int nt = wave * 4 + nti;
        if (nti + 1 < 4) __builtin_prefetch((const void*)&w1f[((nt + 1) * 8) * 32 + lane], 0, 0);
        v8f acc[4] = {{}, {}, {}, {}};
#pragma unroll
        for (int kt = 0; kt < 8; ++kt) {
            v16h b = w1f[(nt * 8 + kt) * 32 + lane];
#pragma unroll
            for (int m = 0; m < 4; ++m) {
                v16h a = load_a_frag(sX + m * 16 * NLD, NLD, kt * 32, lane);
                acc[m] = __builtin_amdgcn_wmma_f32_16x16x32_f16(false, a, false, b, (short)0, acc[m], false, false);
            }
        }
        int col   = nt * 16 + (lane & 15);
        int rbase = (lane & 16) ? 8 : 0;
        float bv  = bias1[col];
#pragma unroll
        for (int m = 0; m < 4; ++m)
#pragma unroll
            for (int g = 0; g < 8; ++g)
                sH[(m * 16 + rbase + g) * NLD + col] = (_Float16)gelu_exact(acc[m][g] + bv);
    }
    __syncthreads();

    // GEMM2: 8 nt over 4 waves -> 2 each
    const v16h* w2f = (const v16h*)w2p;
    for (int nti = 0; nti < 2; ++nti) {
        int nt = wave * 2 + nti;
        v8f acc[4] = {{}, {}, {}, {}};
#pragma unroll
        for (int kt = 0; kt < 8; ++kt) {
            v16h b = w2f[(nt * 8 + kt) * 32 + lane];
#pragma unroll
            for (int m = 0; m < 4; ++m) {
                v16h a = load_a_frag(sH + m * 16 * NLD, NLD, kt * 32, lane);
                acc[m] = __builtin_amdgcn_wmma_f32_16x16x32_f16(false, a, false, b, (short)0, acc[m], false, false);
            }
        }
        int col   = nt * 16 + (lane & 15);
        int rbase = (lane & 16) ? 8 : 0;
        float bv  = bias2[col];
#pragma unroll
        for (int m = 0; m < 4; ++m)
#pragma unroll
            for (int g = 0; g < 8; ++g) {
                long n = n0 + m * 16 + rbase + g;
                if (n < nNodes)
                    node_out[n * 128 + col] = node[n * 128 + col] + (acc[m][g] + bv);
            }
    }
}

// ---------------------------------------------------------------------------
extern "C" void kernel_launch(void* const* d_in, const int* in_sizes, int n_in,
                              void* d_out, int out_size, void* d_ws, size_t ws_size,
                              hipStream_t stream) {
    const float* node    = (const float*)d_in[0];
    const float* edge    = (const float*)d_in[1];
    const int*   edgeIdx = (const int*)d_in[2];
    const int*   rows    = (const int*)d_in[3];
    const int*   cols    = (const int*)d_in[4];
    const float* vals    = (const float*)d_in[5];
    const float* n1g     = (const float*)d_in[6];
    const float* n1b     = (const float*)d_in[7];
    const float* n2g     = (const float*)d_in[8];
    const float* n2b     = (const float*)d_in[9];
    const float* e_w1    = (const float*)d_in[10];
    const float* e_b1    = (const float*)d_in[11];
    const float* e_w2    = (const float*)d_in[12];
    const float* e_b2    = (const float*)d_in[13];
    const float* n_w1    = (const float*)d_in[14];
    const float* n_b1    = (const float*)d_in[15];
    const float* n_w2    = (const float*)d_in[16];
    const float* n_b2    = (const float*)d_in[17];

    const int nN  = in_sizes[0] / 128;   // 50000
    const int nE  = in_sizes[1] / 128;   // 250000
    const int nnz = in_sizes[3];         // 500000

    // workspace layout (halves)
    _Float16* w1e = (_Float16*)d_ws;                 // 384*384
    _Float16* w2e = w1e + 384 * 384;                 // 384*128
    _Float16* w1n = w2e + 384 * 128;                 // 256*256
    _Float16* w2n = w1n + 256 * 256;                 // 256*128
    _Float16* Eh  = w2n + 256 * 128;                 // nE*128 f16
    _Float16* agg = Eh + (size_t)nE * 128;           // nN*128 f16

    float* node_out = (float*)d_out;
    float* edge_out = node_out + (size_t)nN * 128;

    // 1) pack weights f32 -> f16 WMMA B-fragments
    pack_weights_kernel<<<256, 256, 0, stream>>>(e_w1, w1e, 384, 384);
    pack_weights_kernel<<<128, 256, 0, stream>>>(e_w2, w2e, 384, 128);
    pack_weights_kernel<<<128, 256, 0, stream>>>(n_w1, w1n, 256, 256);
    pack_weights_kernel<<<64, 256, 0, stream>>>(n_w2, w2n, 256, 128);

    // 2) zero fp16 aggregation buffer
    long aggWords = (long)nN * 128 / 2;
    zero_u32_kernel<<<1024, 256, 0, stream>>>((uint32_t*)agg, aggWords);

    // 3) edge MLP (WMMA, 64-row workgroup M-tile)
    int eBlocks = (nE + 63) / 64;
    edge_mlp_kernel<<<eBlocks, 128, 0, stream>>>(node, edge, edgeIdx, n1g, n1b,
                                                 w1e, e_b1, w2e, e_b2,
                                                 Eh, edge_out, nE);

    // 4) fp16 packed-atomic scatter-add
    long sThreads = (long)nnz * 64;
    int  sBlocks  = (int)((sThreads + 255) / 256);
    scatter_kernel<<<sBlocks, 256, 0, stream>>>(Eh, rows, cols, vals, agg, nnz);

    // 5) node MLP (WMMA, 64-row workgroup M-tile) + residual
    int nBlocks = (nN + 63) / 64;
    node_mlp_kernel<<<nBlocks, 128, 0, stream>>>(node, agg, n2g, n2b,
                                                 w1n, n_b1, w2n, n_b2,
                                                 node_out, nN);
}
